// GCN_1864015806687
// MI455X (gfx1250) — compile-verified, matrix operand order checked by reading, not run on hardware
//
#include <hip/hip_runtime.h>

// ---------------------------------------------------------------------------
// GCN forward for gfx1250 (MI455X): bf16 WMMA GEMMs (double-buffered LDS,
// async global->LDS staging via GLOBAL_LOAD_ASYNC_TO_LDS_B128) + atomic
// edge aggregation.
// ---------------------------------------------------------------------------

typedef __attribute__((ext_vector_type(16))) __bf16 v16bf;
typedef __attribute__((ext_vector_type(8)))  float  v8f;

#define N_NODES 16384
#define N_EDGES 65536

#define BM 128
#define BN 128
#define BK 32
#define LDT 40   // padded LDS row stride (bf16 elems) to break bank conflicts

#if defined(__gfx1250__) &&                                                    \
    __has_builtin(__builtin_amdgcn_global_load_async_to_lds_b128) &&           \
    __has_builtin(__builtin_amdgcn_s_wait_asynccnt)
#define USE_ASYNC_COPY 1
// Builtin signature (from hipcc diagnostic): params are pointers to
// `int __attribute__((vector_size(16)))` in AS1 (global) / AS3 (LDS).
typedef int v4i_ __attribute__((vector_size(16)));
typedef __attribute__((address_space(1))) v4i_ GV4;  // global 16B vector
typedef __attribute__((address_space(3))) v4i_ LV4;  // LDS 16B vector
#else
#define USE_ASYNC_COPY 0
#endif

union FragU { v16bf v; unsigned int u[8]; };

// Build a 16x32 bf16 WMMA fragment from an LDS tile (row-major, stride LDT).
// ISA layout (16-bit A/B, wave32): lanes 0-15 -> row, elems = K 0..7,16..23;
// lanes 16-31 -> row, elems = K 8..15,24..31.
__device__ __forceinline__ v16bf load_frag_lds(const __bf16* base, int lane) {
  const int half = (lane >> 4) & 1;
  const int row  = lane & 15;
  const __bf16* r = base + row * LDT + half * 8;
  FragU f;
#pragma unroll
  for (int p = 0; p < 8; ++p) {
    const int k = ((p & 3) << 1) + ((p >> 2) << 4);  // 0,2,4,6,16,18,20,22
    f.u[p] = *reinterpret_cast<const unsigned int*>(r + k);
  }
  return f.v;
}

// C[M,F] = A[M,K] (bf16) x Bt[F,K]^T (bf16), f32 accumulate.
// Block: 256 threads (8 wave32), tile 128x128x32, double-buffered LDS.
__global__ __launch_bounds__(256)
void k_gemm_bf16(const __bf16* __restrict__ A, const __bf16* __restrict__ Bt,
                 float* __restrict__ C, int K, int F) {
  __shared__ __bf16 sA[2][BM * LDT];
  __shared__ __bf16 sB[2][BN * LDT];

  const int tid  = threadIdx.x;
  const int lane = tid & 31;
  const int wave = tid >> 5;
  const int m0 = blockIdx.y * BM;
  const int n0 = blockIdx.x * BN;
  const int wr = (wave >> 2) * 64;   // wave row offset: 0 / 64
  const int wc = (wave & 3) * 32;    // wave col offset: 0..96

  v8f acc[4][2] = {};

  const int ld_row = tid >> 2;       // 0..63
  const int ld_col = (tid & 3) * 8;  // 0,8,16,24 (bf16 elems -> 16B chunks)

#if USE_ASYNC_COPY
  // Async DMA of one 128x32 A tile + 128x32 B tile into LDS buffer `buf`.
  auto issue_tile = [&](int kt, int buf) {
#pragma unroll
    for (int h = 0; h < 2; ++h) {
      const int row = ld_row + h * 64;
      __builtin_amdgcn_global_load_async_to_lds_b128(
          (GV4*)(A + (size_t)(m0 + row) * K + kt + ld_col),
          (LV4*)(&sA[buf][row * LDT + ld_col]), 0, 0);
      const int nrow = n0 + row;
      if (nrow < F) {
        __builtin_amdgcn_global_load_async_to_lds_b128(
            (GV4*)(Bt + (size_t)nrow * K + kt + ld_col),
            (LV4*)(&sB[buf][row * LDT + ld_col]), 0, 0);
      } else {  // ragged F (e.g. 1000): zero-fill pad rows
        *reinterpret_cast<uint4*>(&sB[buf][row * LDT + ld_col]) =
            make_uint4(0u, 0u, 0u, 0u);
      }
    }
  };
  issue_tile(0, 0);
  __builtin_amdgcn_s_wait_asynccnt(0);
  __syncthreads();
#else
  {  // sync stage of tile 0
#pragma unroll
    for (int h = 0; h < 2; ++h) {
      const int row = ld_row + h * 64;
      *reinterpret_cast<uint4*>(&sA[0][row * LDT + ld_col]) =
          *reinterpret_cast<const uint4*>(A + (size_t)(m0 + row) * K + ld_col);
      const int nrow = n0 + row;
      uint4 vb = make_uint4(0u, 0u, 0u, 0u);
      if (nrow < F)
        vb = *reinterpret_cast<const uint4*>(Bt + (size_t)nrow * K + ld_col);
      *reinterpret_cast<uint4*>(&sB[0][row * LDT + ld_col]) = vb;
    }
  }
  __syncthreads();
#endif

  for (int kt = 0; kt < K; kt += BK) {
    const int  cur  = (kt / BK) & 1;
    const int  nxt  = cur ^ 1;
    const bool more = (kt + BK) < K;

    // ---- issue next-tile transfers (overlap with compute below) ----
#if USE_ASYNC_COPY
    if (more) issue_tile(kt + BK, nxt);
#else
    uint4 ra[2], rb[2];
    if (more) {
#pragma unroll
      for (int h = 0; h < 2; ++h) {
        const int row = ld_row + h * 64;
        ra[h] = *reinterpret_cast<const uint4*>(
            A + (size_t)(m0 + row) * K + kt + BK + ld_col);
        const int nrow = n0 + row;
        rb[h] = make_uint4(0u, 0u, 0u, 0u);
        if (nrow < F)
          rb[h] = *reinterpret_cast<const uint4*>(
              Bt + (size_t)nrow * K + kt + BK + ld_col);
      }
      __builtin_prefetch(A + (size_t)(m0 + ld_row) * K + kt + 2 * BK, 0, 3);
    }
#endif

    // ---- compute from current buffer ----
    v16bf aF[4], bF[2];
#pragma unroll
    for (int i = 0; i < 4; ++i)
      aF[i] = load_frag_lds(&sA[cur][(wr + i * 16) * LDT], lane);
#pragma unroll
    for (int j = 0; j < 2; ++j)
      bF[j] = load_frag_lds(&sB[cur][(wc + j * 16) * LDT], lane);

#pragma unroll
    for (int i = 0; i < 4; ++i)
#pragma unroll
      for (int j = 0; j < 2; ++j)
        acc[i][j] = __builtin_amdgcn_wmma_f32_16x16x32_bf16(
            false, aF[i], false, bF[j], (short)0, acc[i][j], false, false);

    // ---- commit next tile; one barrier per K-step ----
#if USE_ASYNC_COPY
    if (more) __builtin_amdgcn_s_wait_asynccnt(0);
#else
    if (more) {
#pragma unroll
      for (int h = 0; h < 2; ++h) {
        const int row = ld_row + h * 64;
        *reinterpret_cast<uint4*>(&sA[nxt][row * LDT + ld_col]) = ra[h];
        *reinterpret_cast<uint4*>(&sB[nxt][row * LDT + ld_col]) = rb[h];
      }
    }
#endif
    __syncthreads();
  }

  // C/D layout: lane 0-15 -> N=lane, M=vgpr r; lane 16-31 -> N=lane-16, M=r+8
  const int rsel = (lane >> 4) * 8;
  const int ncol = lane & 15;
#pragma unroll
  for (int i = 0; i < 4; ++i)
#pragma unroll
    for (int j = 0; j < 2; ++j) {
      const int n = n0 + wc + j * 16 + ncol;
      if (n < F) {
#pragma unroll
        for (int r = 0; r < 8; ++r) {
          const int m = m0 + wr + i * 16 + r + rsel;
          C[(size_t)m * F + n] = acc[i][j][r];
        }
      }
    }
}

// ---------------- small helper kernels ----------------

__global__ void k_fill_deg(float* deg) {
  const int i = blockIdx.x * blockDim.x + threadIdx.x;
  if (i < N_NODES) deg[i] = 1.0f;  // self-loop weight
}

__global__ void k_deg_accum(float* deg, const int* __restrict__ dst,
                            const float* __restrict__ ew) {
  const int e = blockIdx.x * blockDim.x + threadIdx.x;
  if (e < N_EDGES) atomicAdd(&deg[dst[e]], ew[e]);
}

__global__ void k_rsqrt_inplace(float* deg) {
  const int i = blockIdx.x * blockDim.x + threadIdx.x;
  if (i < N_NODES) deg[i] = rsqrtf(deg[i]);
}

__global__ void k_edge_coef(const float* __restrict__ dinv,
                            const int* __restrict__ src,
                            const int* __restrict__ dst,
                            const float* __restrict__ ew,
                            float* __restrict__ coef) {
  const int e = blockIdx.x * blockDim.x + threadIdx.x;
  if (e < N_EDGES) coef[e] = dinv[src[e]] * ew[e] * dinv[dst[e]];
}

__global__ void k_cvt_f32_bf16(const float* __restrict__ in,
                               __bf16* __restrict__ out, int n) {
  const int i = blockIdx.x * blockDim.x + threadIdx.x;
  if (i < n) out[i] = (__bf16)in[i];
}

__global__ void k_relu_cvt(const float* __restrict__ in,
                           __bf16* __restrict__ out, int n) {
  const int i = blockIdx.x * blockDim.x + threadIdx.x;
  if (i < n) out[i] = (__bf16)fmaxf(in[i], 0.0f);
}

// W [K,F] f32 -> Wt [F,K] bf16  (grid: x covers F, y = k)
__global__ void k_cvt_wt(const float* __restrict__ W, __bf16* __restrict__ Wt,
                         int K, int F) {
  const int f = blockIdx.x * blockDim.x + threadIdx.x;
  const int k = blockIdx.y;
  if (f < F) Wt[(size_t)f * K + k] = (__bf16)W[(size_t)k * F + f];
}

// AGG[n,f] = H[n,f] * dinv[n]^2 + bias[f]   (grid: x = n, y covers F)
__global__ void k_agg_init(const float* __restrict__ H,
                           const float* __restrict__ dinv,
                           const float* __restrict__ bias,
                           float* __restrict__ AGG, int F) {
  const int n = blockIdx.x;
  const int f = blockIdx.y * blockDim.x + threadIdx.x;
  if (f < F) {
    const float d = dinv[n];
    AGG[(size_t)n * F + f] = H[(size_t)n * F + f] * d * d + bias[f];
  }
}

// AGG[dst,f] += coef[e] * H[src,f]   (grid: x = e, y covers F)
__global__ void k_edge_scatter(float* __restrict__ AGG,
                               const float* __restrict__ H,
                               const int* __restrict__ src,
                               const int* __restrict__ dst,
                               const float* __restrict__ coef, int F) {
  const int e = blockIdx.x;
  const int f = blockIdx.y * blockDim.x + threadIdx.x;
  if (f < F) {
    const int s = src[e], d = dst[e];
    const float c = coef[e];
    atomicAdd(&AGG[(size_t)d * F + f], c * H[(size_t)s * F + f]);
  }
}

// out[n,f] = H[n,f] + bias[f]
__global__ void k_bias_out(const float* __restrict__ H,
                           const float* __restrict__ bias,
                           float* __restrict__ out, int F) {
  const int n = blockIdx.x;
  const int f = blockIdx.y * blockDim.x + threadIdx.x;
  if (f < F) out[(size_t)n * F + f] = H[(size_t)n * F + f] + bias[f];
}

// ---------------------------------------------------------------------------

extern "C" void kernel_launch(void* const* d_in, const int* in_sizes, int n_in,
                              void* d_out, int out_size, void* d_ws,
                              size_t ws_size, hipStream_t stream) {
  (void)in_sizes; (void)n_in; (void)out_size; (void)ws_size;

  const float* x    = (const float*)d_in[0];
  const int*   esrc = (const int*)d_in[1];
  const int*   edst = (const int*)d_in[2];
  const float* ew   = (const float*)d_in[3];
  const float* Wl[5] = {(const float*)d_in[4], (const float*)d_in[6],
                        (const float*)d_in[8], (const float*)d_in[10],
                        (const float*)d_in[12]};
  const float* bl[5] = {(const float*)d_in[5], (const float*)d_in[7],
                        (const float*)d_in[9], (const float*)d_in[11],
                        (const float*)d_in[13]};
  float* out = (float*)d_out;

  // workspace carve-out
  char* ws = (char*)d_ws;
  size_t off = 0;
  auto carve = [&](size_t bytes) -> void* {
    off = (off + 255) & ~(size_t)255;
    void* p = ws + off;
    off += bytes;
    return p;
  };
  float*  H    = (float*)carve((size_t)N_NODES * 4096 * 4);   // GEMM out
  float*  AGG  = (float*)carve((size_t)N_NODES * 4096 * 4);   // aggregation
  __bf16* Xb   = (__bf16*)carve((size_t)N_NODES * 4096 * 2);  // act (bf16)
  __bf16* Wt   = (__bf16*)carve((size_t)4096 * 4096 * 2);     // W^T (bf16)
  float*  dinv = (float*)carve((size_t)N_NODES * 4);
  float*  coef = (float*)carve((size_t)N_EDGES * 4);

  const int T = 256;

  // degree -> dinv -> per-edge coefficients (recomputed every call)
  k_fill_deg<<<(N_NODES + T - 1) / T, T, 0, stream>>>(dinv);
  k_deg_accum<<<(N_EDGES + T - 1) / T, T, 0, stream>>>(dinv, edst, ew);
  k_rsqrt_inplace<<<(N_NODES + T - 1) / T, T, 0, stream>>>(dinv);
  k_edge_coef<<<(N_EDGES + T - 1) / T, T, 0, stream>>>(dinv, esrc, edst, ew,
                                                       coef);
  // x -> bf16
  {
    const int n = N_NODES * 512;
    k_cvt_f32_bf16<<<(n + T - 1) / T, T, 0, stream>>>(x, Xb, n);
  }

  const int dims[6] = {512, 1024, 2048, 4096, 2048, 1000};

  // 4 GCN conv layers with ReLU
  for (int l = 0; l < 4; ++l) {
    const int K = dims[l], F = dims[l + 1];
    k_cvt_wt<<<dim3((F + T - 1) / T, K), T, 0, stream>>>(Wl[l], Wt, K, F);
    k_gemm_bf16<<<dim3((F + BN - 1) / BN, N_NODES / BM), T, 0, stream>>>(
        Xb, Wt, H, K, F);
    k_agg_init<<<dim3(N_NODES, (F + T - 1) / T), T, 0, stream>>>(H, dinv,
                                                                 bl[l], AGG, F);
    k_edge_scatter<<<dim3(N_EDGES, (F + T - 1) / T), T, 0, stream>>>(
        AGG, H, esrc, edst, coef, F);
    const int n = N_NODES * F;
    k_relu_cvt<<<(n + T - 1) / T, T, 0, stream>>>(AGG, Xb, n);
  }

  // output layer: plain dense + bias (no graph conv per reference)
  {
    const int K = dims[4], F = dims[5];
    k_cvt_wt<<<dim3((F + T - 1) / T, K), T, 0, stream>>>(Wl[4], Wt, K, F);
    k_gemm_bf16<<<dim3((F + BN - 1) / BN, N_NODES / BM), T, 0, stream>>>(
        Xb, Wt, H, K, F);
    k_bias_out<<<dim3(N_NODES, (F + T - 1) / T), T, 0, stream>>>(H, bl[4], out,
                                                                 F);
  }
}
